// AwesomeGRU_9216999817259
// MI455X (gfx1250) — matrix operand order repaired and checked by reading, not compiled, not run on hardware
//
#include <hip/hip_runtime.h>

// ---------------------------------------------------------------------------
// GRU with reset mask on MI455X (gfx1250, wave32, WMMA).
//   gi GEMM:  (SEQ*B=32768) x (3U=3072) x (DIM=1024)  -> bf16 WMMA, f32 acc
//   scan:     512 sequential step kernels; W_hh (6MB bf16) L2-resident.
//             h carried in TWO forms: f32 slab in d_out (exact z*h path) and
//             a bf16 double buffer for the next step's WMMA A operand, so the
//             f32->bf16 conversion happens once per element, not once per
//             reading workgroup. Reset mask ({0,1} per batch row) is applied
//             as a per-lane cndmask select on the loaded A fragment.
// ---------------------------------------------------------------------------

#define SEQ   512
#define BATCH 64
#define DIM   1024
#define UNITS 1024
#define G3    (3 * UNITS)   // 3072

typedef __attribute__((ext_vector_type(16))) __bf16 v16bf;
typedef __attribute__((ext_vector_type(8)))  float  v8f;

static __device__ __forceinline__ unsigned short f32_bf16(float f) {
  unsigned int u = __float_as_uint(f);
  u += 0x7FFFu + ((u >> 16) & 1u);          // round-to-nearest-even
  return (unsigned short)(u >> 16);
}
static __device__ __forceinline__ float bf16_f32(unsigned short s) {
  return __uint_as_float(((unsigned int)s) << 16);
}

// ---- f32 -> bf16 bulk conversion, 4 elements/thread --------------------------
__global__ void __launch_bounds__(256)
cvt_f32_bf16_x4(const float4* __restrict__ in, uint2* __restrict__ out, int n4) {
  int i = blockIdx.x * 256 + threadIdx.x;
  if (i < n4) {
    float4 v = in[i];
    uint2 o;
    o.x = (unsigned)f32_bf16(v.x) | ((unsigned)f32_bf16(v.y) << 16);
    o.y = (unsigned)f32_bf16(v.z) | ((unsigned)f32_bf16(v.w) << 16);
    out[i] = o;
  }
}

// ---- WMMA fragment loaders (ISA 7.12.2 layouts) ------------------------------
// A 16x32 bf16: lane&15 -> row M; half-wave picks K {0-7,16-23} vs {8-15,24-31}.
static __device__ __forceinline__ v16bf load_a_bf16(
    const unsigned short* __restrict__ base, int lda, int lane) {
  int l = lane & 15, hi = lane >> 4;
  const unsigned short* row = base + l * lda + hi * 8;
  union { uint4 u[2]; v16bf v; } f;
  f.u[0] = *(const uint4*)(row);        // K = k0 + hi*8 .. +7
  f.u[1] = *(const uint4*)(row + 16);   // K = k0 + 16 + hi*8 .. +7
  return f.v;
}

// Same A layout, with the {0,1} reset mask folded in as a lane-level select.
static __device__ __forceinline__ v16bf load_a_bf16_masked(
    const unsigned short* __restrict__ base, int lda, int lane, bool keep) {
  int l = lane & 15, hi = lane >> 4;
  const unsigned short* row = base + l * lda + hi * 8;
  uint4 t0 = *(const uint4*)(row);
  uint4 t1 = *(const uint4*)(row + 16);
  uint4 z; z.x = 0u; z.y = 0u; z.z = 0u; z.w = 0u;
  union { uint4 u[2]; v16bf v; } f;
  f.u[0] = keep ? t0 : z;               // v_cndmask x4
  f.u[1] = keep ? t1 : z;
  return f.v;
}

// B 32x16 bf16 from weight rows (column n of B == row n0+n of W, contiguous K):
// lane&15 -> column N; half-wave picks K {0-7,16-23} vs {8-15,24-31}.
static __device__ __forceinline__ v16bf load_b_w(
    const unsigned short* __restrict__ base, int ldw, int lane) {
  int n = lane & 15, hi = lane >> 4;
  const unsigned short* row = base + n * ldw + hi * 8;
  union { uint4 u[2]; v16bf v; } f;
  f.u[0] = *(const uint4*)(row);
  f.u[1] = *(const uint4*)(row + 16);
  return f.v;
}

// ---- gi = x @ W_ih^T + b_ih, output bf16 ------------------------------------
// Each wave computes a 32(M) x 64(N) block: 2 M-tiles x 4 N-tiles, K=1024.
__global__ void __launch_bounds__(128)
gi_gemm(const unsigned short* __restrict__ X,     // (SEQ*B) x DIM bf16
        const unsigned short* __restrict__ Wih,   // 3U x DIM bf16
        const float*          __restrict__ bih,   // 3U
        unsigned short*       __restrict__ GI) {  // (SEQ*B) x 3U bf16
  const int lane = threadIdx.x & 31;
  const int wave = blockIdx.x * 4 + (threadIdx.x >> 5);
  const int NB = G3 / 64;                 // 48  (N fastest -> x rows L2-hot)
  const int nb = wave % NB, mb = wave / NB;
  const int m0 = mb * 32, n0 = nb * 64;

  v8f c[2][4];
#pragma unroll
  for (int mi = 0; mi < 2; ++mi)
#pragma unroll
    for (int ni = 0; ni < 4; ++ni) c[mi][ni] = {};

  for (int k0 = 0; k0 < DIM; k0 += 32) {
    v16bf a0 = load_a_bf16(X + (size_t)(m0 +  0) * DIM + k0, DIM, lane);
    v16bf a1 = load_a_bf16(X + (size_t)(m0 + 16) * DIM + k0, DIM, lane);
#pragma unroll
    for (int ni = 0; ni < 4; ++ni) {
      v16bf b = load_b_w(Wih + (size_t)(n0 + ni * 16) * DIM + k0, DIM, lane);
      c[0][ni] = __builtin_amdgcn_wmma_f32_16x16x32_bf16(
          false, a0, false, b, (short)0, c[0][ni], false, false);
      c[1][ni] = __builtin_amdgcn_wmma_f32_16x16x32_bf16(
          false, a1, false, b, (short)0, c[1][ni], false, false);
    }
  }

  const int cl = lane & 15, hi = lane >> 4;
#pragma unroll
  for (int ni = 0; ni < 4; ++ni) {
    int col = n0 + ni * 16 + cl;
    float bias = bih[col];
#pragma unroll
    for (int mi = 0; mi < 2; ++mi) {
#pragma unroll
      for (int i = 0; i < 8; ++i) {
        int row = m0 + mi * 16 + i + hi * 8;       // C layout: VGPR i -> M=i / 8+i
        GI[(size_t)row * G3 + col] = f32_bf16(c[mi][ni][i] + bias);
      }
    }
  }
}

// ---- one GRU scan step -------------------------------------------------------
// grid = UNITS/16 blocks, 4 waves; wave -> batch tile m0, block -> unit slice u0.
// gh = (h*mask) @ W_hh^T + b_hh via 3 WMMA accumulators (r,z,n gates), then
// fused gate math; h_new written to out[s] (f32) and Hb_out (bf16, A operand
// for step s+1).
__global__ void __launch_bounds__(128)
gru_step(const unsigned short* __restrict__ Hb_in,  // B x UNITS bf16 (h_{s-1})
         const float*          __restrict__ Hprev,  // B x UNITS f32  (h_{s-1})
         const unsigned short* __restrict__ GIs,    // B x 3U bf16 (step s)
         const int*            __restrict__ Rst,    // B (step s)
         const unsigned short* __restrict__ Whh,    // 3U x UNITS bf16
         const float*          __restrict__ bhh,    // 3U
         float*                __restrict__ Hout,   // B x UNITS f32 (out[s])
         unsigned short*       __restrict__ Hb_out) // B x UNITS bf16
{
  const int lane = threadIdx.x & 31;
  const int m0 = (threadIdx.x >> 5) * 16;
  const int u0 = blockIdx.x * 16;
  const bool keep = (Rst[m0 + (lane & 15)] == 0);   // mask row for this lane's A

  v8f cr = {}, cz = {}, cn = {};
  const unsigned short* Wr = Whh + (size_t)(0 * UNITS + u0) * UNITS;
  const unsigned short* Wz = Whh + (size_t)(1 * UNITS + u0) * UNITS;
  const unsigned short* Wn = Whh + (size_t)(2 * UNITS + u0) * UNITS;

#pragma unroll 2
  for (int k0 = 0; k0 < UNITS; k0 += 32) {
    v16bf a  = load_a_bf16_masked(Hb_in + (size_t)m0 * UNITS + k0, UNITS, lane, keep);
    v16bf br = load_b_w(Wr + k0, UNITS, lane);
    v16bf bz = load_b_w(Wz + k0, UNITS, lane);
    v16bf bn = load_b_w(Wn + k0, UNITS, lane);
    cr = __builtin_amdgcn_wmma_f32_16x16x32_bf16(false, a, false, br, (short)0, cr, false, false);
    cz = __builtin_amdgcn_wmma_f32_16x16x32_bf16(false, a, false, bz, (short)0, cz, false, false);
    cn = __builtin_amdgcn_wmma_f32_16x16x32_bf16(false, a, false, bn, (short)0, cn, false, false);
  }

  const int cl = lane & 15, hi = lane >> 4;
  const int col = u0 + cl;
  const float br_ = bhh[col], bz_ = bhh[UNITS + col], bn_ = bhh[2 * UNITS + col];
#pragma unroll
  for (int i = 0; i < 8; ++i) {
    int row = m0 + i + hi * 8;
    float gir = bf16_f32(GIs[(size_t)row * G3 + col]);
    float giz = bf16_f32(GIs[(size_t)row * G3 + UNITS + col]);
    float gin = bf16_f32(GIs[(size_t)row * G3 + 2 * UNITS + col]);
    float ghr = cr[i] + br_;
    float ghz = cz[i] + bz_;
    float ghn = cn[i] + bn_;
    float r = 1.0f / (1.0f + __expf(-(gir + ghr)));
    float z = 1.0f / (1.0f + __expf(-(giz + ghz)));
    float n = tanhf(gin + r * ghn);
    float hm = (Rst[row] == 0) ? Hprev[(size_t)row * UNITS + col] : 0.0f;
    float hn = (1.0f - z) * n + z * hm;
    Hout[(size_t)row * UNITS + col] = hn;
    Hb_out[(size_t)row * UNITS + col] = f32_bf16(hn);
  }
}

// ---------------------------------------------------------------------------
extern "C" void kernel_launch(void* const* d_in, const int* in_sizes, int n_in,
                              void* d_out, int out_size, void* d_ws, size_t ws_size,
                              hipStream_t stream) {
  const float* x     = (const float*)d_in[0];  // SEQ*B*DIM
  const float* h0    = (const float*)d_in[1];  // B*UNITS
  const int*   reset = (const int*)  d_in[2];  // SEQ*B
  const float* W_ih  = (const float*)d_in[3];  // 3U*DIM
  const float* W_hh  = (const float*)d_in[4];  // 3U*UNITS
  const float* b_ih  = (const float*)d_in[5];  // 3U
  const float* b_hh  = (const float*)d_in[6];  // 3U
  float* out = (float*)d_out;                  // SEQ*B*UNITS

  // workspace layout (bf16 copies + bf16 gi + bf16 h double buffer), ~281 MB
  char* ws = (char*)d_ws;
  size_t off = 0;
  unsigned short* xb  = (unsigned short*)(ws + off); off += (size_t)SEQ * BATCH * DIM * 2;  // 64 MB
  unsigned short* wib = (unsigned short*)(ws + off); off += (size_t)G3 * DIM * 2;           // 6 MB
  unsigned short* whb = (unsigned short*)(ws + off); off += (size_t)G3 * UNITS * 2;         // 6 MB
  unsigned short* gib = (unsigned short*)(ws + off); off += (size_t)SEQ * BATCH * G3 * 2;   // 192 MB
  unsigned short* hb0 = (unsigned short*)(ws + off); off += (size_t)BATCH * UNITS * 2;      // 128 KB
  unsigned short* hb1 = (unsigned short*)(ws + off);

  // 1) f32 -> bf16 conversions (x, W_ih, W_hh, h0)
  {
    int n4 = (SEQ * BATCH * DIM) / 4;
    cvt_f32_bf16_x4<<<(n4 + 255) / 256, 256, 0, stream>>>(
        (const float4*)x, (uint2*)xb, n4);
    n4 = (G3 * DIM) / 4;
    cvt_f32_bf16_x4<<<(n4 + 255) / 256, 256, 0, stream>>>(
        (const float4*)W_ih, (uint2*)wib, n4);
    n4 = (G3 * UNITS) / 4;
    cvt_f32_bf16_x4<<<(n4 + 255) / 256, 256, 0, stream>>>(
        (const float4*)W_hh, (uint2*)whb, n4);
    n4 = (BATCH * UNITS) / 4;
    cvt_f32_bf16_x4<<<(n4 + 255) / 256, 256, 0, stream>>>(
        (const float4*)h0, (uint2*)hb0, n4);
  }

  // 2) gi = x @ W_ih^T + b_ih  (one big WMMA GEMM)
  {
    int waves = ((SEQ * BATCH) / 32) * (G3 / 64);  // 49152
    gi_gemm<<<waves / 4, 128, 0, stream>>>(xb, wib, b_ih, gib);
  }

  // 3) sequential scan: 512 step kernels, stream order = inter-step barrier
  for (int s = 0; s < SEQ; ++s) {
    const unsigned short* hb_in  = (s & 1) ? hb1 : hb0;
    unsigned short*       hb_out = (s & 1) ? hb0 : hb1;
    const float* hp = (s == 0) ? h0 : out + (size_t)(s - 1) * BATCH * UNITS;
    gru_step<<<UNITS / 16, 128, 0, stream>>>(
        hb_in, hp,
        gib + (size_t)s * BATCH * G3,
        reset + (size_t)s * BATCH,
        whb, b_hh,
        out + (size_t)s * BATCH * UNITS,
        hb_out);
  }
}
